// TemporalMamba_14388140441605
// MI455X (gfx1250) — matrix operand order repaired and checked by reading, not compiled, not run on hardware
//
#include <hip/hip_runtime.h>
#include <cstdint>

// ---- problem constants --------------------------------------------------
#define BB      8
#define CC      128
#define HW      4096            // 64*64
#define NTOK    32768           // BB*HW
#define DM      256             // 2*CC
#define DI      512             // 2*DM
#define DSTATE  16
#define DTR     16

// ---- WMMA fragment types (wave32, gfx1250) ------------------------------
typedef __attribute__((ext_vector_type(16))) __bf16 v16bf;
typedef __attribute__((ext_vector_type(8)))  float  v8f;

union Frag16 {
    v16bf    v;
    uint4    q[2];
    uint32_t u[8];
    uint16_t s[16];
};

__device__ __forceinline__ v8f vzero8() {
    v8f z = {0.f,0.f,0.f,0.f,0.f,0.f,0.f,0.f};
    return z;
}

__device__ __forceinline__ v8f wmma_bf16(v16bf a, v16bf b, v8f c) {
    // D = A(16x32 bf16) * B(32x16 bf16) + C(16x16 f32)
    return __builtin_amdgcn_wmma_f32_16x16x32_bf16(
        /*neg_a=*/false, a, /*neg_b=*/false, b,
        /*c_mod=*/(short)0, c, /*reuse_a=*/false, /*reuse_b=*/false);
}

// ---- bf16 helpers (RNE, bit-exact, no reliance on __bf16 arithmetic) ----
__device__ __forceinline__ uint16_t f2bf(float f) {
    uint32_t u = __builtin_bit_cast(uint32_t, f);
    u += 0x7FFFu + ((u >> 16) & 1u);
    return (uint16_t)(u >> 16);
}
__device__ __forceinline__ float bf2f(uint16_t h) {
    uint32_t u = ((uint32_t)h) << 16;
    return __builtin_bit_cast(float, u);
}
__device__ __forceinline__ float silu(float x) {
    return x / (1.f + __expf(-x));
}

// =========================================================================
// Prep: f32 -> bf16 weight conversion
// =========================================================================
__global__ void k_cvt_bf16(const float* __restrict__ src,
                           uint16_t* __restrict__ dst, int n) {
    int i = blockIdx.x * blockDim.x + threadIdx.x;
    if (i < n) dst[i] = f2bf(src[i]);
}

// Prep: Wc[c][n] = sum_d proj_w[c][d] * out_proj_w[d][n]   (128 x 512, bf16)
__global__ void k_fuse_w(const float* __restrict__ Wp,    // (128,256)
                         const float* __restrict__ Wo,    // (256,512)
                         uint16_t* __restrict__ Wc) {
    int idx = blockIdx.x * blockDim.x + threadIdx.x;      // 65536 threads
    int c = idx >> 9, n = idx & 511;
    float acc = 0.f;
    #pragma unroll 4
    for (int d = 0; d < DM; ++d)
        acc = fmaf(Wp[c * DM + d], Wo[d * DI + n], acc);
    Wc[idx] = f2bf(acc);
}

// =========================================================================
// K1: xz = S @ in_proj_w.T   (S built from f1/f2 in LDS), epilogue:
//     Xs = silu(x*conv_tap + conv_b) [bf16], Zs = silu(z) [bf16]
// 512 blocks x 256 threads; block = 64 tokens; 8 waves = 4 tok-tiles x 2 strips
// =========================================================================
#define S_ROW_U32 129   // 64 rows x 129 u32 (=258 bf16, padded): conflict-free
__global__ __launch_bounds__(256) void k1_inproj(
    const float* __restrict__ f1, const float* __restrict__ f2,
    const uint16_t* __restrict__ Wa,          // in_proj_w bf16 (1024,256)
    const float* __restrict__ conv_w,         // (512,1,4)
    const float* __restrict__ conv_b,         // (512,)
    uint16_t* __restrict__ Xs,                // (N,512) bf16
    uint16_t* __restrict__ Zs)                // (N,512) bf16
{
    __shared__ uint32_t Stile[64 * S_ROW_U32];

    const int tid = threadIdx.x;
    const int t0  = blockIdx.x * 64;
    const int b   = t0 >> 12;                 // token block never straddles b
    const int p0  = t0 & (HW - 1);

    // Build S tile: token i, channel-pair c -> u32 {f1:lo(k=2c), f2:hi(k=2c+1)}
    for (int idx = tid; idx < 64 * CC; idx += 256) {
        int c = idx >> 6;                     // 0..127
        int i = idx & 63;                     // token in tile (coalesced)
        size_t gaddr = (size_t)b * (CC * HW) + (size_t)c * HW + (p0 + i);
        uint32_t lo = f2bf(f1[gaddr]);
        uint32_t hi = f2bf(f2[gaddr]);
        Stile[i * S_ROW_U32 + c] = lo | (hi << 16);
    }
    __syncthreads();

    const int lane = tid & 31;
    const int wave = tid >> 5;
    const int tt   = wave & 3;                // token tile 0..3
    const int strip= wave >> 2;               // 0..1
    const int m    = lane & 15;
    const int g    = lane >> 4;
    const int rowb = (tt * 16 + m) * S_ROW_U32;

    for (int nblk = 0; nblk < 4; ++nblk) {
        const int n_base = nblk * 256 + strip * 128;
        v8f acc[8];
        #pragma unroll
        for (int j = 0; j < 8; ++j) acc[j] = vzero8();

        for (int k0 = 0; k0 < DM; k0 += 32) {
            Frag16 A;
            const int c1 = (k0 + g * 16) >> 1;    // u32 col of chunk 1
            #pragma unroll
            for (int j = 0; j < 4; ++j) {
                A.u[j]     = Stile[rowb + c1 + j];
                A.u[4 + j] = Stile[rowb + c1 + 4 + j];
            }
            #pragma unroll
            for (int j = 0; j < 8; ++j) {
                const uint16_t* wp =
                    Wa + (size_t)(n_base + j * 16 + m) * DM + k0 + g * 16;
                Frag16 Bf;
                Bf.q[0] = *(const uint4*)(wp);
                Bf.q[1] = *(const uint4*)(wp + 8);
                acc[j] = wmma_bf16(A.v, Bf.v, acc[j]);
            }
        }

        // epilogue (whole 128-col strip is entirely x-half or z-half)
        const bool isX = (n_base < DI);
        #pragma unroll
        for (int j = 0; j < 8; ++j) {
            const int n  = n_base + j * 16 + m;
            const int nn = isX ? n : (n - DI);
            const float cw = isX ? conv_w[nn * 4 + 3] : 0.f;
            const float cb = isX ? conv_b[nn] : 0.f;
            uint16_t* dst = isX ? Xs : Zs;
            #pragma unroll
            for (int v = 0; v < 8; ++v) {
                const int t = t0 + tt * 16 + g * 8 + v;
                float val = acc[j][v];
                float r = isX ? silu(val * cw + cb) : silu(val);
                dst[(size_t)t * DI + nn] = f2bf(r);
            }
        }
    }
}

// =========================================================================
// K2: x_dbl = Xs @ x_proj_w.T (48 cols) -> bc, dt GEMM (K=16 zero-padded
//     to 32), softplus, elementwise combine; Y overwrites Zs in place.
// 512 blocks x 128 threads; wave = 16 tokens
// =========================================================================
__global__ __launch_bounds__(128) void k2_scan(
    const uint16_t* __restrict__ Xs,          // (N,512) bf16
    uint16_t* ZY,                             // in: silu(z); out: y (bf16)
    const uint16_t* __restrict__ Wxp,         // x_proj_w bf16 (48,512)
    const uint16_t* __restrict__ Wdt,         // dt_proj_w bf16 (512,16)
    const float* __restrict__ dt_b,           // (512,)
    const float* __restrict__ Dp)             // (512,)
{
    __shared__ float xdbl[4][16][52];         // 48 cols padded to 52
    __shared__ float bcs[4][16];

    const int tid  = threadIdx.x;
    const int lane = tid & 31;
    const int w    = tid >> 5;
    const int m    = lane & 15;
    const int g    = lane >> 4;
    const int t0   = blockIdx.x * 64 + w * 16;
    const int trow = t0 + m;

    // --- x_proj GEMM: 16 tokens x 48 outputs, K = 512 ---
    v8f acc[3];
    #pragma unroll
    for (int j = 0; j < 3; ++j) acc[j] = vzero8();

    for (int k0 = 0; k0 < DI; k0 += 32) {
        Frag16 A;
        const uint16_t* xp = Xs + (size_t)trow * DI + k0 + g * 16;
        A.q[0] = *(const uint4*)(xp);
        A.q[1] = *(const uint4*)(xp + 8);
        #pragma unroll
        for (int j = 0; j < 3; ++j) {
            const uint16_t* wp = Wxp + (size_t)(j * 16 + m) * DI + k0 + g * 16;
            Frag16 Bf;
            Bf.q[0] = *(const uint4*)(wp);
            Bf.q[1] = *(const uint4*)(wp + 8);
            acc[j] = wmma_bf16(A.v, Bf.v, acc[j]);
        }
    }
    #pragma unroll
    for (int j = 0; j < 3; ++j)
        #pragma unroll
        for (int v = 0; v < 8; ++v)
            xdbl[w][g * 8 + v][j * 16 + m] = acc[j][v];
    __syncthreads();

    // bc[row] = dot(Bm, Cm)
    if (g == 0) {
        float s = 0.f;
        #pragma unroll
        for (int q = 0; q < DSTATE; ++q)
            s = fmaf(xdbl[w][m][DTR + q], xdbl[w][m][DTR + DSTATE + q], s);
        bcs[w][m] = s;
    }
    __syncthreads();

    // A fragment for dt GEMM: rows = tokens, K = dt_rank (16) zero-padded
    Frag16 Adt;
    #pragma unroll
    for (int i = 0; i < 16; ++i) {
        int k = g * 16 + i;
        Adt.s[i] = (k < DTR) ? f2bf(xdbl[w][m][k]) : (uint16_t)0;
    }

    // dt GEMM over 512 outputs, fused softplus + elementwise epilogue
    for (int nb = 0; nb < 32; ++nb) {
        const int n0 = nb * 16;
        Frag16 Bf;
        if (g == 0) {  // K=0..15 real, K=16..31 zero (lanes 16..31)
            const uint16_t* wp = Wdt + (size_t)(n0 + m) * DTR;
            Bf.q[0] = *(const uint4*)(wp);
            Bf.q[1] = *(const uint4*)(wp + 8);
        } else {
            #pragma unroll
            for (int j = 0; j < 8; ++j) Bf.u[j] = 0u;
        }
        v8f d = wmma_bf16(Adt.v, Bf.v, vzero8());

        const int n   = n0 + m;
        const float db = dt_b[n];
        const float dp = Dp[n];
        #pragma unroll
        for (int v = 0; v < 8; ++v) {
            const int row = g * 8 + v;
            const size_t off = (size_t)(t0 + row) * DI + n;
            float pre = d[v] + db;
            float dtv = (pre > 20.f) ? pre : log1pf(__expf(pre));
            float xv  = bf2f(Xs[off]);
            float zv  = bf2f(ZY[off]);
            float y   = xv * (dtv * bcs[w][row] + dp) * zv;
            ZY[off] = f2bf(y);
        }
    }
}

// =========================================================================
// K3: out = Y @ Wc.T + proj_b, LDS transpose for coalesced BCHW stores
// 512 blocks x 256 threads; 8 waves = 4 tok-tiles x 2 col strips (64 each)
// =========================================================================
#define O_ROW_F32 129
__global__ __launch_bounds__(256) void k3_out(
    const uint16_t* __restrict__ Ybf,         // (N,512) bf16
    const uint16_t* __restrict__ Wc,          // fused (128,512) bf16
    const float* __restrict__ proj_b,         // (128,)
    float* __restrict__ out)                  // (B,C,H,W) f32
{
    __shared__ float Ot[64 * O_ROW_F32];

    const int tid  = threadIdx.x;
    const int lane = tid & 31;
    const int wave = tid >> 5;
    const int m    = lane & 15;
    const int g    = lane >> 4;
    const int tt   = wave >> 1;
    const int strip= wave & 1;
    const int t0   = blockIdx.x * 64;
    const int trow = t0 + tt * 16 + m;

    v8f acc[4];
    #pragma unroll
    for (int j = 0; j < 4; ++j) acc[j] = vzero8();

    for (int k0 = 0; k0 < DI; k0 += 32) {
        Frag16 A;
        const uint16_t* yp = Ybf + (size_t)trow * DI + k0 + g * 16;
        A.q[0] = *(const uint4*)(yp);
        A.q[1] = *(const uint4*)(yp + 8);
        #pragma unroll
        for (int j = 0; j < 4; ++j) {
            const uint16_t* wp =
                Wc + (size_t)(strip * 64 + j * 16 + m) * DI + k0 + g * 16;
            Frag16 Bf;
            Bf.q[0] = *(const uint4*)(wp);
            Bf.q[1] = *(const uint4*)(wp + 8);
            acc[j] = wmma_bf16(A.v, Bf.v, acc[j]);
        }
    }

    #pragma unroll
    for (int j = 0; j < 4; ++j) {
        const int c = strip * 64 + j * 16 + m;
        const float bias = proj_b[c];
        #pragma unroll
        for (int v = 0; v < 8; ++v) {
            const int row = tt * 16 + g * 8 + v;
            Ot[row * O_ROW_F32 + c] = acc[j][v] + bias;
        }
    }
    __syncthreads();

    const int b  = t0 >> 12;
    const int p0 = t0 & (HW - 1);
    for (int idx = tid; idx < 64 * CC; idx += 256) {
        const int c = idx >> 6;
        const int i = idx & 63;                // coalesced along p
        out[(size_t)b * (CC * HW) + (size_t)c * HW + p0 + i] =
            Ot[i * O_ROW_F32 + c];
    }
}

// =========================================================================
// Host launch
// =========================================================================
extern "C" void kernel_launch(void* const* d_in, const int* in_sizes, int n_in,
                              void* d_out, int out_size, void* d_ws, size_t ws_size,
                              hipStream_t stream) {
    (void)in_sizes; (void)n_in; (void)out_size; (void)ws_size;

    const float* f1        = (const float*)d_in[0];
    const float* f2        = (const float*)d_in[1];
    const float* in_proj_w = (const float*)d_in[2];   // (1024,256)
    const float* conv_w    = (const float*)d_in[3];   // (512,1,4)
    const float* conv_b    = (const float*)d_in[4];   // (512,)
    const float* x_proj_w  = (const float*)d_in[5];   // (48,512)
    const float* dt_proj_w = (const float*)d_in[6];   // (512,16)
    const float* dt_proj_b = (const float*)d_in[7];   // (512,)
    /* d_in[8] = A_log: unused by the reference computation */
    const float* Dp        = (const float*)d_in[9];   // (512,)
    const float* out_proj_w= (const float*)d_in[10];  // (256,512)
    const float* proj_w    = (const float*)d_in[11];  // (128,256)
    const float* proj_b    = (const float*)d_in[12];  // (128,)
    float* out = (float*)d_out;

    char* ws = (char*)d_ws;
    uint16_t* Wa  = (uint16_t*)ws; ws += (size_t)1024 * 256 * 2;   // 512 KB
    uint16_t* Wxp = (uint16_t*)ws; ws += (size_t)48  * 512 * 2;    //  48 KB
    uint16_t* Wdt = (uint16_t*)ws; ws += (size_t)512 * 16  * 2;    //  16 KB
    uint16_t* Wc  = (uint16_t*)ws; ws += (size_t)128 * 512 * 2;    // 128 KB
    uint16_t* Xs  = (uint16_t*)ws; ws += (size_t)NTOK * DI * 2;    //  32 MB
    uint16_t* ZY  = (uint16_t*)ws; ws += (size_t)NTOK * DI * 2;    //  32 MB

    // weight prep
    k_cvt_bf16<<<(1024 * 256 + 255) / 256, 256, 0, stream>>>(in_proj_w, Wa, 1024 * 256);
    k_cvt_bf16<<<(48 * 512 + 255) / 256, 256, 0, stream>>>(x_proj_w, Wxp, 48 * 512);
    k_cvt_bf16<<<(512 * 16 + 255) / 256, 256, 0, stream>>>(dt_proj_w, Wdt, 512 * 16);
    k_fuse_w<<<(128 * 512) / 256, 256, 0, stream>>>(proj_w, out_proj_w, Wc);

    // pipeline
    k1_inproj<<<NTOK / 64, 256, 0, stream>>>(f1, f2, Wa, conv_w, conv_b, Xs, ZY);
    k2_scan  <<<NTOK / 64, 128, 0, stream>>>(Xs, ZY, Wxp, Wdt, dt_proj_b, Dp);
    k3_out   <<<NTOK / 64, 256, 0, stream>>>(ZY, Wc, proj_b, out);
}
